// KANConv2d_71047349010847
// MI455X (gfx1250) — matrix operand order repaired and checked by reading, not compiled, not run on hardware
//
#include <hip/hip_runtime.h>

typedef __attribute__((ext_vector_type(16))) _Float16 v16h;
typedef __attribute__((ext_vector_type(8)))  _Float16 v8h;
typedef __attribute__((ext_vector_type(8)))  float    v8f;
typedef __attribute__((__vector_size__(16))) int      v4i;   // builtin's pointee type

#define CIN   64
#define HW    56
#define OC    128
#define FIN   576        // CIN * 3 * 3
#define GAUG  9          // 1 (silu) + 8 spline bases
#define KAUG  5184       // FIN * GAUG
#define NTOT  25088      // 8 * 56 * 56
#define BM    64         // rows (pixels) per block
#define IBLK  32         // features per i-block
#define NIB   18         // FIN / IBLK

#if __has_builtin(__builtin_amdgcn_global_load_async_to_lds_b128) && \
    __has_builtin(__builtin_amdgcn_s_wait_asynccnt)
#define USE_ASYNC 1
#else
#define USE_ASYNC 0
#endif

// ---------------------------------------------------------------------------
// Prep: fuse base_weight / spline_weight*scaler into W_aug[OC][KAUG] (f16),
// K-ordering is g-major: kidx = g*FIN + i  (g=0 -> silu weights).
// ---------------------------------------------------------------------------
__global__ __launch_bounds__(256) void kan_prep_waug(
    const float* __restrict__ base_w,     // [OC][FIN]
    const float* __restrict__ spline_w,   // [OC][FIN][8]
    const float* __restrict__ scaler,     // [OC][FIN]
    _Float16* __restrict__ waug)          // [OC][KAUG]
{
    int idx = blockIdx.x * 256 + threadIdx.x;
    if (idx >= OC * KAUG) return;
    int o  = idx / KAUG;
    int kk = idx - o * KAUG;
    int g  = kk / FIN;
    int i  = kk - g * FIN;
    float v;
    if (g == 0) v = base_w[o * FIN + i];
    else        v = spline_w[(o * FIN + i) * 8 + (g - 1)] * scaler[o * FIN + i];
    waug[idx] = (_Float16)v;
}

// Load one 16x32 f16 A-fragment from an LDS tile (row stride 32 halves).
// Lanes 0-15: row=lane K={0..7,16..23}; lanes 16-31: row=lane-16 K={8..15,24..31}.
__device__ __forceinline__ v16h ld_fragA(const _Float16* tile, int lane) {
    int r = lane & 15, sub = (lane >> 4) & 1;
    const _Float16* p = tile + r * 32 + sub * 8;
    v8h lo = *(const v8h*)(p);
    v8h hi = *(const v8h*)(p + 16);
    return __builtin_shufflevector(lo, hi, 0,1,2,3,4,5,6,7,8,9,10,11,12,13,14,15);
}

// Load one 32x16 f16 B-fragment from an LDS tile stored [N][K] (stride 32).
// Lanes 0-15: col=lane K={0..7,16..23}; lanes 16-31: col=lane-16 K={8..15,24..31}.
__device__ __forceinline__ v16h ld_fragB(const _Float16* tile, int lane) {
    int r = lane & 15, sub = (lane >> 4) & 1;
    const _Float16* p = tile + r * 32 + sub * 8;
    v8h lo = *(const v8h*)(p);
    v8h hi = *(const v8h*)(p + 16);
    return __builtin_shufflevector(lo, hi, 0,1,2,3,4,5,6,7,8,9,10,11,12,13,14,15);
}

// Copy one 128x32-half W_aug tile into LDS (per-thread: 32 bytes).
__device__ __forceinline__ void load_btile(const _Float16* __restrict__ waug,
                                           _Float16* dst, int g, int ib, int tid) {
    int o = tid >> 1, half = tid & 1;
    const _Float16* src = waug + o * KAUG + g * FIN + ib * IBLK + half * 16;
    _Float16* d = dst + o * IBLK + half * 16;
#if USE_ASYNC
    // CDNA5 async copy: global -> LDS, no VGPR staging, tracked by ASYNCcnt.
    __builtin_amdgcn_global_load_async_to_lds_b128((v4i*)src, (v4i*)d, 0, 0);
    __builtin_amdgcn_global_load_async_to_lds_b128((v4i*)(src + 8), (v4i*)(d + 8), 0, 0);
#else
    v8h b0 = *(const v8h*)(src);
    v8h b1 = *(const v8h*)(src + 8);
    *(v8h*)(d)     = b0;
    *(v8h*)(d + 8) = b1;
    if (g + 1 < GAUG)
        __builtin_prefetch(waug + o * KAUG + (g + 1) * FIN + ib * IBLK + half * 16, 0, 1);
#endif
}

__device__ __forceinline__ void wait_async() {
#if USE_ASYNC
    __builtin_amdgcn_s_wait_asynccnt(0);
#endif
}

// ---------------------------------------------------------------------------
// Fused KAN-conv GEMM: out[N, OC] = A_aug[N, KAUG] x W_aug^T, A built on the
// fly from im2col(x) (silu + cardinal cubic B-spline bases, uniform grid
// h=0.4 on [-1,1], 8 bases).
// ---------------------------------------------------------------------------
__global__ __launch_bounds__(256) void kan_conv_wmma(
    const float*    __restrict__ x,       // [8][64][56][56]
    const _Float16* __restrict__ waug,    // [OC][KAUG]
    float*          __restrict__ out)     // [8][OC][56][56]
{
    __shared__ __align__(16) _Float16 sAug[GAUG * BM * IBLK]; // 36 KB
    __shared__ __align__(16) _Float16 sB0[OC * IBLK];         // 8 KB (double-
    __shared__ __align__(16) _Float16 sB1[OC * IBLK];         // 8 KB  buffered)

    const int tid  = threadIdx.x;
    const int lane = tid & 31;
    const int wid  = tid >> 5;      // 8 waves
    const int wm   = wid & 1;       // 2 row-tiles of 32
    const int wn   = wid >> 1;      // 4 col-tiles of 32
    const int n0   = blockIdx.x * BM;

    v8f acc[2][2] = {};

    for (int ib = 0; ib < NIB; ++ib) {
        __syncthreads();                   // sAug WAR + sB0 WAR from prev ib
        load_btile(waug, sB0, 0, ib, tid); // g=0 tile copies under phase 1

        // ---- Phase 1: build augmented activation tile [g][row][k] ----
        #pragma unroll
        for (int t = 0; t < 8; ++t) {
            int p   = t * 256 + tid;      // 0..2047
            int row = p & (BM - 1);       // fast index -> coalesced x loads
            int kk  = p >> 6;             // 0..31
            int n = n0 + row;
            int i = ib * IBLK + kk;
            int b  = n / 3136; int hw = n - b * 3136;
            int oh = hw / HW;  int ow = hw - oh * HW;
            int c  = i / 9;    int r9 = i - c * 9;
            int kh = r9 / 3;   int kw = r9 - kh * 3;
            int ih = oh + kh - 1, iw = ow + kw - 1;
            float xv = 0.0f;
            if (ih >= 0 && ih < HW && iw >= 0 && iw < HW)
                xv = x[((b * CIN + c) * HW + ih) * HW + iw];

            // silu
            float sig = 1.0f / (1.0f + __expf(-xv));
            float w0 = xv * sig;

            // cardinal cubic B-spline: knots t_j = -1 + 0.4*(j-3), j=0..11
            float u  = (xv + 1.0f) * 2.5f + 3.0f;   // knot-space coordinate
            float fj = floorf(u);
            int   j  = (int)fj;
            float f  = u - fj;
            float f2 = f * f, f3 = f2 * f, omf = 1.0f - f;
            bool  ind = (u >= 0.0f) && (u < 11.0f);
            float c0 = ind ? omf * omf * omf * (1.0f / 6.0f) : 0.0f;
            float c1 = ind ? (3.0f * f3 - 6.0f * f2 + 4.0f) * (1.0f / 6.0f) : 0.0f;
            float c2 = ind ? (-3.0f * f3 + 3.0f * f2 + 3.0f * f + 1.0f) * (1.0f / 6.0f) : 0.0f;
            float c3 = ind ? f3 * (1.0f / 6.0f) : 0.0f;
            int g0 = j - 3;   // first active basis index

            _Float16* dst = sAug + row * IBLK + kk;
            dst[0] = (_Float16)w0;
            #pragma unroll
            for (int g = 0; g < 8; ++g) {   // static index -> no scratch
                int d = g - g0;
                float wv = 0.0f;
                wv = (d == 0) ? c0 : wv;
                wv = (d == 1) ? c1 : wv;
                wv = (d == 2) ? c2 : wv;
                wv = (d == 3) ? c3 : wv;
                dst[(g + 1) * (BM * IBLK)] = (_Float16)wv;
            }
        }

        // ---- Phase 2: 9 WMMA k-chunks, B tiles double-buffered ----
        for (int g = 0; g < GAUG; ++g) {
            _Float16* cur = (g & 1) ? sB1 : sB0;
            _Float16* nxt = (g & 1) ? sB0 : sB1;
            wait_async();     // our async copy of tile g complete
            __syncthreads();  // tile g + sAug visible; buffer `nxt` free (WAR)
            if (g + 1 < GAUG)
                load_btile(waug, nxt, g + 1, ib, tid);  // overlaps WMMAs below

            const _Float16* At = sAug + g * (BM * IBLK) + (wm * 32) * IBLK;
            const _Float16* Bt = cur + (wn * 32) * IBLK;
            v16h a0 = ld_fragA(At,             lane);
            v16h a1 = ld_fragA(At + 16 * IBLK, lane);
            v16h b0 = ld_fragB(Bt,             lane);
            v16h b1 = ld_fragB(Bt + 16 * IBLK, lane);

            acc[0][0] = __builtin_amdgcn_wmma_f32_16x16x32_f16(false, a0, false, b0, (short)0, acc[0][0], false, false);
            acc[0][1] = __builtin_amdgcn_wmma_f32_16x16x32_f16(false, a0, false, b1, (short)0, acc[0][1], false, false);
            acc[1][0] = __builtin_amdgcn_wmma_f32_16x16x32_f16(false, a1, false, b0, (short)0, acc[1][0], false, false);
            acc[1][1] = __builtin_amdgcn_wmma_f32_16x16x32_f16(false, a1, false, b1, (short)0, acc[1][1], false, false);
        }
    }

    // ---- Epilogue: scatter C into NCHW output ----
    #pragma unroll
    for (int fi = 0; fi < 2; ++fi) {
        #pragma unroll
        for (int fj = 0; fj < 2; ++fj) {
            #pragma unroll
            for (int r = 0; r < 8; ++r) {
                int mrow = r + ((lane >> 4) & 1) * 8;
                int n = n0 + wm * 32 + fi * 16 + mrow;
                int o = wn * 32 + fj * 16 + (lane & 15);
                int b  = n / 3136;
                int hw = n - b * 3136;
                out[(b * OC + o) * 3136 + hw] = acc[fi][fj][r];
            }
        }
    }
}

extern "C" void kernel_launch(void* const* d_in, const int* in_sizes, int n_in,
                              void* d_out, int out_size, void* d_ws, size_t ws_size,
                              hipStream_t stream) {
    const float* x  = (const float*)d_in[0];   // [8][64][56][56]
    const float* bw = (const float*)d_in[1];   // [128][576]
    const float* sw = (const float*)d_in[2];   // [128][576][8]
    const float* sc = (const float*)d_in[3];   // [128][576]
    float* out      = (float*)d_out;           // [8][128][56][56]
    _Float16* waug  = (_Float16*)d_ws;         // [128][5184] f16 = 1.33 MB

    kan_prep_waug<<<(OC * KAUG + 255) / 256, 256, 0, stream>>>(bw, sw, sc, waug);
    kan_conv_wmma<<<NTOT / BM, 256, 0, stream>>>(x, waug, out);
}
